// MultiHeadAttention_87290915324513
// MI455X (gfx1250) — matrix operand order repaired and checked by reading, not compiled, not run on hardware
//
#include <hip/hip_runtime.h>
#include <cstdint>

// ---------------------------------------------------------------------------
// MHA for MI455X (gfx1250, wave32, WMMA + TDM).
// ~206 GFLOP dense math, flash-style softmax (no S*S tensor). All matmuls on
// v_wmma_f32_16x16x32_bf16 (fp32 accumulate). Per-head K/V (2 MB bf16) are
// L2-resident (192 MB), so attention fragments stream from global memory.
// Final projection stages its bf16 A-tiles with the Tensor Data Mover.
// ---------------------------------------------------------------------------

#define SEQ 2048
#define DIM 512
#define NB  2
#define NH  8
#define ED  512
#define HE  (NH*ED)     // 4096
#define NEGV (-1e9f)

typedef __attribute__((ext_vector_type(8)))  float    v8f;
typedef __attribute__((ext_vector_type(4)))  float    v4f;
typedef __attribute__((ext_vector_type(16))) __bf16   v16bf;
typedef __attribute__((ext_vector_type(8)))  __bf16   v8bf;
typedef __attribute__((ext_vector_type(4)))  __bf16   v4bf;
typedef __attribute__((ext_vector_type(4)))  unsigned v4u;
typedef __attribute__((ext_vector_type(8)))  int      v8i;
typedef __attribute__((ext_vector_type(4)))  int      v4i;

union Frag16 { v16bf v; v8bf h[2]; };

__device__ inline v8f zero8() {
  v8f r = {0.f,0.f,0.f,0.f,0.f,0.f,0.f,0.f};
  return r;
}

__device__ inline v8f wmma_bf16(v16bf a, v16bf b, v8f c) {
  // D = A(16x32 bf16) * B(32x16 bf16) + C(16x16 f32)
  return __builtin_amdgcn_wmma_f32_16x16x32_bf16(false, a, false, b,
                                                 (short)0, c, false, false);
}

// ---------------------------------------------------------------------------
// Tiled GEMM: C[z] = A[z] * B[z] + bias[z]
//   A: [M,K] row-major, fp32 (ABF=0) or bf16 (ABF=1), ld = K
//   B: [K,N] row-major fp32, converted to bf16 while staging, ld = N
//   C: bf16 (OUTF32=0) or fp32 (OUTF32=1); TRANSC writes C^T with ld = M
//   TDM=1 (requires ABF): A tile staged by tensor_load_to_lds DMA.
// Block tile 128x128, K-step 32; 8 waves, each owns 2(M)x4(N) WMMA tiles.
// ---------------------------------------------------------------------------
template<bool ABF, bool OUTF32, bool TRANSC, bool TDM>
__global__ __launch_bounds__(256)
void gemm_wmma(const float* __restrict__ Af, const __bf16* __restrict__ Ab,
               const float* __restrict__ Bf,
               const float* __restrict__ bias,
               __bf16* __restrict__ Cb, float* __restrict__ Cf,
               int M, int N, int K, int zdiv, int zmod,
               long azs, long bzs, long biass, long czs) {
  __shared__ __align__(64) __bf16 As[128*32];   // [m][k]  row stride 32
  __shared__ __align__(64) __bf16 Bs[128*32];   // [n][k]  (B stored transposed)

  const int z = blockIdx.z;
  const float*  A32 = ABF ? nullptr : Af + (size_t)(z / zdiv) * azs;
  const __bf16* A16 = ABF ? Ab + (size_t)(z / zdiv) * azs : nullptr;
  const float*  Bg  = Bf + (size_t)(z % zmod) * bzs;
  const float*  bg  = bias + (size_t)(z % zmod) * biass;
  const size_t  coff = (size_t)z * czs;

  const int tid  = threadIdx.x;
  const int wave = tid >> 5;
  const int lane = tid & 31;
  const int lm   = lane & 15;     // N (or M for A-frag) index within tile
  const int hi   = lane >> 4;     // lane-half select
  const int wM   = wave >> 1;     // 0..3
  const int wN   = wave & 1;      // 0..1
  const int m0   = blockIdx.y * 128;
  const int n0   = blockIdx.x * 128;

  v8f acc[2][4];
#pragma unroll
  for (int i = 0; i < 2; ++i)
#pragma unroll
    for (int j = 0; j < 4; ++j) acc[i][j] = zero8();

  for (int k0 = 0; k0 < K; k0 += 32) {
    // ---- stage A tile --------------------------------------------------
    if constexpr (TDM) {
      // Tensor Data Mover: 2D tile (32 x 128 rows) of the bf16 tensor
      // [M x K] -> LDS, rows contiguous = exactly As's [m][k] layout.
      if (tid < 32) {
        unsigned ldsa = (unsigned)(size_t)(&As[0]);
        unsigned long long ga =
            (unsigned long long)(size_t)(const void*)&A16[(size_t)m0 * K + k0];
        v4u g0 = { 1u,                                   // count=1 (valid D#)
                   ldsa,                                  // lds_addr
                   (unsigned)(ga & 0xffffffffu),
                   (unsigned)((ga >> 32) & 0x01ffffffu) | (2u << 30) }; // type=2
        unsigned kd = (unsigned)K, md = (unsigned)M;
        v8i g1 = { (int)(1u << 16),                       // data_size = 2B
                   (int)((kd & 0xffffu) << 16),           // tensor_dim0 lo16
                   (int)(((kd >> 16) & 0xffffu) | ((md & 0xffffu) << 16)),
                   (int)(((md >> 16) & 0xffffu) | (32u << 16)),  // tile_dim0=32
                   (int)128,                              // tile_dim1=128
                   (int)kd,                               // dim0_stride lo32
                   0, 0 };
        v4i gz = {0,0,0,0};
#if __clang_major__ >= 23
        v8i gz8 = {0,0,0,0,0,0,0,0};
        __builtin_amdgcn_tensor_load_to_lds(g0, g1, gz, gz, gz8, 0);
#else
        __builtin_amdgcn_tensor_load_to_lds(g0, g1, gz, gz, 0);
#endif
        __builtin_amdgcn_s_wait_tensorcnt(0);
      }
    } else if constexpr (ABF) {
#pragma unroll
      for (int it = 0; it < 4; ++it) {
        int flat = (tid + it*256) * 4;
        int mm = flat >> 5, kk = flat & 31;
        *(v4bf*)&As[mm*32 + kk] =
            *(const v4bf*)&A16[(size_t)(m0 + mm) * K + k0 + kk];
      }
    } else {
#pragma unroll
      for (int it = 0; it < 4; ++it) {
        int flat = (tid + it*256) * 4;
        int mm = flat >> 5, kk = flat & 31;
        v4f a = *(const v4f*)&A32[(size_t)(m0 + mm) * K + k0 + kk];
        v4bf cv = { (__bf16)a.x, (__bf16)a.y, (__bf16)a.z, (__bf16)a.w };
        *(v4bf*)&As[mm*32 + kk] = cv;
      }
    }
    // ---- stage B tile transposed: Bs[n][k] = B[k][n] -------------------
#pragma unroll
    for (int it = 0; it < 4; ++it) {
      int flat = tid + it*256;          // group of 4 along k
      int nn = flat & 127, kk = (flat >> 7) * 4;
      const float* bp = &Bg[(size_t)(k0 + kk) * N + n0 + nn];
      v4bf cv = { (__bf16)bp[0], (__bf16)bp[(size_t)N],
                  (__bf16)bp[(size_t)2*N], (__bf16)bp[(size_t)3*N] };
      *(v4bf*)&Bs[nn*32 + kk] = cv;
    }
    __syncthreads();

    // ---- fragments, k = 16*hi + s in A and B ------------------------------------
    Frag16 af[2];
#pragma unroll
    for (int i = 0; i < 2; ++i) {
      const __bf16* p = &As[(wM*32 + i*16 + lm) * 32 + hi*16];
      af[i].h[0] = *(const v8bf*)p;
      af[i].h[1] = *(const v8bf*)(p + 8);
    }
    v16bf bfr[4];
#pragma unroll
    for (int j = 0; j < 4; ++j)
      bfr[j] = *(const v16bf*)&Bs[(wN*64 + j*16 + lm) * 32 + hi*16];

#pragma unroll
    for (int i = 0; i < 2; ++i)
#pragma unroll
      for (int j = 0; j < 4; ++j)
        acc[i][j] = wmma_bf16(af[i].v, bfr[j], acc[i][j]);
    __syncthreads();
  }

  // ---- writeback (C layout: lane = N, vgpr r -> row r + 8*hi) ----------
#pragma unroll
  for (int i = 0; i < 2; ++i)
#pragma unroll
    for (int j = 0; j < 4; ++j)
#pragma unroll
      for (int r = 0; r < 8; ++r) {
        int row = m0 + wM*32 + i*16 + r + 8*hi;
        int col = n0 + wN*64 + j*16 + lm;
        float v = acc[i][j][r] + bg[col];
        if constexpr (OUTF32) {
          Cf[coff + (size_t)row * N + col] = v;
        } else if constexpr (TRANSC) {
          Cb[coff + (size_t)col * M + row] = (__bf16)v;
        } else {
          Cb[coff + (size_t)row * N + col] = (__bf16)v;
        }
      }
}

// ---------------------------------------------------------------------------
// Attention pass 1: streaming online-softmax stats (row max m, row sum l).
// Block = 8 waves, each wave owns 16 query rows; sweeps all S keys in 16-wide
// tiles. QK^T via WMMA; B-fragment of K^T is just row-major K (contiguous).
// Row reductions via shfl_xor within the 16-lane N-groups of the C layout.
// ---------------------------------------------------------------------------
__global__ __launch_bounds__(256)
void attn_stats(const __bf16* __restrict__ qb, const __bf16* __restrict__ kbuf,
                const uint8_t* __restrict__ mask,
                float* __restrict__ mstat, float* __restrict__ lstat) {
  const int z = blockIdx.y;               // b*NH + h
  const int b = z / NH;
  const __bf16* q  = qb   + (size_t)z * SEQ * ED;
  const __bf16* kk = kbuf + (size_t)z * SEQ * ED;
  const uint8_t* mk = mask + (size_t)b * SEQ * SEQ;

  const int tid  = threadIdx.x;
  const int wave = tid >> 5;
  const int lane = tid & 31;
  const int lm   = lane & 15;
  const int hi   = lane >> 4;
  const int qrow0 = blockIdx.x * 128 + wave * 16;

  // Hoist all 16 Q A-fragments (full E=512 contraction) into registers.
  Frag16 qf[16];
  const __bf16* qp = q + (size_t)(qrow0 + lm) * ED;
#pragma unroll
  for (int eo = 0; eo < 16; ++eo) {
    qf[eo].h[0] = *(const v8bf*)(qp + eo*32 + hi*16);
    qf[eo].h[1] = *(const v8bf*)(qp + eo*32 + 8 + hi*16);
  }

  float mrun[8], lrun[8];
#pragma unroll
  for (int r = 0; r < 8; ++r) { mrun[r] = -3.0e38f; lrun[r] = 0.f; }

  for (int kt = 0; kt < SEQ/16; ++kt) {
    v8f s = zero8();
    const __bf16* kp = kk + (size_t)(kt*16 + lm) * ED + hi*16;
#pragma unroll
    for (int eo = 0; eo < 16; ++eo) {
      v16bf bfr = *(const v16bf*)(kp + eo*32);
      s = wmma_bf16(qf[eo].v, bfr, s);
    }
    const int key = kt*16 + lm;
#pragma unroll
    for (int r = 0; r < 8; ++r) {
      int row = qrow0 + 8*hi + r;
      float sv = mk[(size_t)row * SEQ + key] ? s[r] : NEGV;
      float tmax = sv;
#pragma unroll
      for (int d = 1; d < 16; d <<= 1)
        tmax = fmaxf(tmax, __shfl_xor(tmax, d, 32));
      float mnew = fmaxf(mrun[r], tmax);
      float p = __expf(sv - mnew);
#pragma unroll
      for (int d = 1; d < 16; d <<= 1)
        p += __shfl_xor(p, d, 32);
      lrun[r] = lrun[r] * __expf(mrun[r] - mnew) + p;
      mrun[r] = mnew;
    }
  }
  if (lm == 0) {
#pragma unroll
    for (int r = 0; r < 8; ++r) {
      int row = qrow0 + 8*hi + r;
      mstat[(size_t)z * SEQ + row] = mrun[r];
      lstat[(size_t)z * SEQ + row] = lrun[r];
    }
  }
}

// ---------------------------------------------------------------------------
// Attention pass 2: out = softmax(QK^T) * V, written into concat layout.
// Wave owns 16 rows x 64 E-cols. Score tiles recomputed (flash), P transposed
// C-layout -> A-layout through a 1 KB per-wave LDS bounce, then P·V WMMA from
// the transposed V^T buffer (unit-stride B fragments).
// ---------------------------------------------------------------------------
__global__ __launch_bounds__(256)
void attn_out(const __bf16* __restrict__ qb, const __bf16* __restrict__ kbuf,
              const __bf16* __restrict__ vT, const uint8_t* __restrict__ mask,
              const float* __restrict__ mstat, const float* __restrict__ lstat,
              __bf16* __restrict__ cat) {
  __shared__ __align__(64) __bf16 P[8][16*32];   // per-wave P tile [m][k]

  const int z = blockIdx.z;
  const int b = z / NH;
  const int h = z % NH;
  const __bf16* q  = qb   + (size_t)z * SEQ * ED;
  const __bf16* kk = kbuf + (size_t)z * SEQ * ED;
  const __bf16* vt = vT   + (size_t)z * ED * SEQ;
  const uint8_t* mk = mask + (size_t)b * SEQ * SEQ;

  const int tid  = threadIdx.x;
  const int wave = tid >> 5;
  const int lane = tid & 31;
  const int lm   = lane & 15;
  const int hi   = lane >> 4;
  const int wM   = wave >> 1;    // 0..3  (row tiles)
  const int wN   = wave & 1;     // 0..1  (col tiles)
  const int qrow0 = blockIdx.y * 64 + wM * 16;
  const int ecol0 = blockIdx.x * 128 + wN * 64;
  __bf16* Pw = P[wave];

  Frag16 qf[16];
  const __bf16* qp = q + (size_t)(qrow0 + lm) * ED;
#pragma unroll
  for (int eo = 0; eo < 16; ++eo) {
    qf[eo].h[0] = *(const v8bf*)(qp + eo*32 + hi*16);
    qf[eo].h[1] = *(const v8bf*)(qp + eo*32 + 8 + hi*16);
  }

  float ms[8], ls[8];
#pragma unroll
  for (int r = 0; r < 8; ++r) {
    ms[r] = mstat[(size_t)z * SEQ + qrow0 + 8*hi + r];
    ls[r] = lstat[(size_t)z * SEQ + qrow0 + 8*hi + r];
  }

  v8f oacc[4];
#pragma unroll
  for (int j = 0; j < 4; ++j) oacc[j] = zero8();

  for (int kb2 = 0; kb2 < SEQ/32; ++kb2) {
#pragma unroll
    for (int t = 0; t < 2; ++t) {
      const int kt = kb2*2 + t;
      v8f s = zero8();
      const __bf16* kp = kk + (size_t)(kt*16 + lm) * ED + hi*16;
#pragma unroll
      for (int eo = 0; eo < 16; ++eo) {
        v16bf bfr = *(const v16bf*)(kp + eo*32);
        s = wmma_bf16(qf[eo].v, bfr, s);
      }
      const int key = kt*16 + lm;
#pragma unroll
      for (int r = 0; r < 8; ++r) {
        int row = qrow0 + 8*hi + r;
        float sv = mk[(size_t)row * SEQ + key] ? s[r] : NEGV;
        float p = __expf(sv - ms[r]);
        Pw[(8*hi + r)*32 + t*16 + lm] = (__bf16)p;  // C-layout -> [m][k] LDS
      }
    }
    // A-fragment of P (LDS ops from the same wave are in order).
    Frag16 pa;
    pa.h[0] = *(const v8bf*)&Pw[lm*32 + hi*16];
    pa.h[1] = *(const v8bf*)&Pw[lm*32 + 8 + hi*16];

#pragma unroll
    for (int j = 0; j < 4; ++j) {
      const __bf16* vp = vt + (size_t)(ecol0 + j*16 + lm) * SEQ + kb2*32 + hi*16;
      v16bf bv = *(const v16bf*)vp;
      oacc[j] = wmma_bf16(pa.v, bv, oacc[j]);
    }
  }

  // Writeback to concat layout [B, S, H*E], normalized by 1/l.
#pragma unroll
  for (int j = 0; j < 4; ++j)
#pragma unroll
    for (int r = 0; r < 8; ++r) {
      int row = qrow0 + 8*hi + r;
      int ec  = ecol0 + j*16 + lm;
      float v = oacc[j][r] / ls[r];
      cat[((size_t)b * SEQ + row) * HE + (size_t)h * ED + ec] = (__bf16)v;
    }
}

// ---------------------------------------------------------------------------
extern "C" void kernel_launch(void* const* d_in, const int* in_sizes, int n_in,
                              void* d_out, int out_size, void* d_ws, size_t ws_size,
                              hipStream_t stream) {
  const float*   q    = (const float*)d_in[0];
  const float*   k    = (const float*)d_in[1];
  const float*   v    = (const float*)d_in[2];
  const uint8_t* mask = (const uint8_t*)d_in[3];
  const float*   Wq   = (const float*)d_in[4];
  const float*   bq   = (const float*)d_in[5];
  const float*   Wk   = (const float*)d_in[6];
  const float*   bk   = (const float*)d_in[7];
  const float*   Wv   = (const float*)d_in[8];
  const float*   bv   = (const float*)d_in[9];
  const float*   Wo   = (const float*)d_in[10];
  const float*   bo   = (const float*)d_in[11];
  float* out = (float*)d_out;

  // Workspace: q,k bf16 [B,H,S,E]; v^T bf16 [B,H,E,S]; cat bf16 [B,S,H*E];
  // softmax stats fp32 [B,H,S] x2.  Total ~134.5 MB.
  const size_t per = (size_t)NB * NH * SEQ * ED;      // 16,777,216 elements
  __bf16* qb  = (__bf16*)d_ws;
  __bf16* kb  = qb + per;
  __bf16* vT  = kb + per;
  __bf16* cat = vT + per;
  float*  mstat = (float*)(cat + per);
  float*  lstat = mstat + (size_t)NB * NH * SEQ;

  dim3 blk(256);

  // Q/K/V projections: per (b,h) GEMM [S,D]x[D,E]+bias.
  dim3 gp(ED/128, SEQ/128, NB*NH);
  gemm_wmma<false,false,false,false><<<gp, blk, 0, stream>>>(
      q, nullptr, Wq, bq, qb, nullptr,
      SEQ, ED, DIM, NH, NH, (long)SEQ*DIM, (long)DIM*ED, ED, (long)SEQ*ED);
  gemm_wmma<false,false,false,false><<<gp, blk, 0, stream>>>(
      k, nullptr, Wk, bk, kb, nullptr,
      SEQ, ED, DIM, NH, NH, (long)SEQ*DIM, (long)DIM*ED, ED, (long)SEQ*ED);
  gemm_wmma<false,false,true,false><<<gp, blk, 0, stream>>>(
      v, nullptr, Wv, bv, vT, nullptr,
      SEQ, ED, DIM, NH, NH, (long)SEQ*DIM, (long)DIM*ED, ED, (long)SEQ*ED);

  // Flash attention: stats pass, then output pass into concat layout.
  attn_stats<<<dim3(SEQ/128, NB*NH), blk, 0, stream>>>(qb, kb, mask, mstat, lstat);
  attn_out<<<dim3(ED/128, SEQ/64, NB*NH), blk, 0, stream>>>(qb, kb, vT, mask,
                                                            mstat, lstat, cat);

  // Output projection: [B*S, H*E] x [H*E, E] + bo -> fp32 d_out.
  // A-tiles staged by the Tensor Data Mover (bf16 source, no conversion).
  gemm_wmma<true,true,false,true><<<dim3(ED/128, (NB*SEQ)/128, 1), blk, 0, stream>>>(
      nullptr, cat, Wo, bo, nullptr, out,
      NB*SEQ, ED, HE, 1, 1, 0, 0, 0, 0);
}
